// GlobalAttention_67216238182741
// MI455X (gfx1250) — matrix operand order repaired and checked by reading, not compiled, not run on hardware
//
#include <hip/hip_runtime.h>
#include <hip/hip_bf16.h>
#include <math.h>

// ---------------------------------------------------------------------------
// Problem constants (reference: B=8, TGT=1024, SRC=1024, DIM=1024)
// ---------------------------------------------------------------------------
#define BATCH 8
#define TGT   1024
#define SRC   1024
#define DIM   1024

typedef __attribute__((ext_vector_type(16))) __bf16 v16bf;
typedef __attribute__((ext_vector_type(8)))  __bf16 v8bf;
typedef __attribute__((ext_vector_type(8)))  float  v8f;

// ---------------------------------------------------------------------------
// f32 -> bf16 conversion, one row per block (no integer division).
// Used to build bf16 operand buffers; `input` is scattered into the right
// half of the pre-concatenated [T, 2*DIM] buffer via dstStride/dstOff.
// ---------------------------------------------------------------------------
__global__ __launch_bounds__(256)
void cvt_f32_bf16_rows(const float* __restrict__ src, __bf16* __restrict__ dst,
                       int rowLen, int dstStride, int dstOff) {
    const size_t r = blockIdx.x;
    const float* s = src + r * (size_t)rowLen;
    __bf16*      d = dst + r * (size_t)dstStride + dstOff;
    for (int c = threadIdx.x; c < rowLen; c += 256)
        d[c] = (__bf16)s[c];
}

// ---------------------------------------------------------------------------
// Fragment loaders, contiguous k order (k = 16*(lane/16) + s) shared by A and B; not the ISA table order
// A 16x32 bf16 (MxK): m = lane%16; lanes 0-15 hold K=0..15,
// lanes 16-31 hold K=16..31 -> two contiguous 16B chunks per lane.
// ---------------------------------------------------------------------------
__device__ __forceinline__
v16bf load_a_frag(const __bf16* __restrict__ A, int lda, int m0, int k0, int lane) {
    int m  = m0 + (lane & 15);
    int kb = (lane < 16) ? 0 : 16;
    const __bf16* p = A + (size_t)m * lda + k0 + kb;
    v8bf lo = *(const v8bf*)(p);        // K = k0+kb .. +7
    v8bf hi = *(const v8bf*)(p + 8);    // K = k0+kb+8 .. +15
    v16bf f;
#pragma unroll
    for (int i = 0; i < 8; ++i) { f[i] = lo[i]; f[i + 8] = hi[i]; }
    return f;
}

// B 32x16 bf16 (KxN), K-contiguous source: B[k][n] = Bm[n*ldb + k].
// n = lane%16; lanes 0-15 hold K=0..15, lanes 16-31 hold K=16..31.
__device__ __forceinline__
v16bf load_b_frag_kc(const __bf16* __restrict__ Bm, int ldb, int n0, int k0, int lane) {
    int n  = n0 + (lane & 15);
    int kb = k0 + ((lane < 16) ? 0 : 16);
    const __bf16* p = Bm + (size_t)n * ldb + kb;
    v8bf lo = *(const v8bf*)(p);
    v8bf hi = *(const v8bf*)(p + 8);
    v16bf f;
#pragma unroll
    for (int i = 0; i < 8; ++i) { f[i] = lo[i]; f[i + 8] = hi[i]; }
    return f;
}

// B 32x16 bf16 (KxN), N-contiguous source: B[k][n] = Bm[k*ldb + n].
__device__ __forceinline__
v16bf load_b_frag_nc(const __bf16* __restrict__ Bm, int ldb, int n0, int k0, int lane) {
    int n  = n0 + (lane & 15);
    int kb = k0 + ((lane < 16) ? 0 : 16);
    v16bf f;
#pragma unroll
    for (int r = 0; r < 8; ++r) {
        f[2 * r]     = Bm[(size_t)(kb + 2 * r)     * ldb + n];
        f[2 * r + 1] = Bm[(size_t)(kb + 2 * r + 1) * ldb + n];
    }
    return f;
}

__device__ __forceinline__ float fast_tanh(float x) {
    float t = __expf(2.0f * x);
    return (t - 1.0f) / (t + 1.0f);
}

// ---------------------------------------------------------------------------
// Batched bf16 WMMA GEMM:  D[b][m][n] = sum_k A[b][m][k] * B(k,n)
// One wave per 32x64 tile: 2 A-frags x 4 B-frags -> 8 WMMAs per K=32 step
// (B traffic amortized over both M halves). All mode flags are template
// parameters so the inner loop is branch-free.
// ---------------------------------------------------------------------------
template <bool KC, bool TANH, bool OUTF32, bool OUTBF>
__global__ __launch_bounds__(32)
void wmma_gemm_bf16(const __bf16* __restrict__ A, long long sA, int lda,
                    const __bf16* __restrict__ Bm, long long sB, int ldb,
                    float* __restrict__ Df32, long long sD32, int ldd32,
                    __bf16* __restrict__ Dbf, long long sDbf, int lddbf,
                    int K) {
    const int lane  = threadIdx.x;          // 32-thread block == one wave32
    const int nTile = blockIdx.x * 64;
    const int mTile = blockIdx.y * 32;
    const int b     = blockIdx.z;

    const __bf16* Ab = A  + (size_t)b * sA;
    const __bf16* Bb = Bm + (size_t)b * sB;

    v8f acc[2][4] = {};

    for (int k0 = 0; k0 < K; k0 += 32) {
        // Branch-free speculative prefetch of the next K-step of A
        // (global_prefetch_b8; dropped harmlessly at the tail).
        {
            int kp = (k0 + 32 < K) ? (k0 + 32) : k0;
            __builtin_prefetch(Ab + (size_t)(mTile + (lane & 15)) * lda + kp, 0, 1);
            __builtin_prefetch(Ab + (size_t)(mTile + 16 + (lane & 15)) * lda + kp, 0, 1);
        }

        const v16bf a0 = load_a_frag(Ab, lda, mTile,      k0, lane);
        const v16bf a1 = load_a_frag(Ab, lda, mTile + 16, k0, lane);

        v16bf bf[4];
#pragma unroll
        for (int j = 0; j < 4; ++j)
            bf[j] = KC ? load_b_frag_kc(Bb, ldb, nTile + 16 * j, k0, lane)
                       : load_b_frag_nc(Bb, ldb, nTile + 16 * j, k0, lane);

#pragma unroll
        for (int j = 0; j < 4; ++j) {
            acc[0][j] = __builtin_amdgcn_wmma_f32_16x16x32_bf16(
                false, a0, false, bf[j], (short)0, acc[0][j], false, false);
            acc[1][j] = __builtin_amdgcn_wmma_f32_16x16x32_bf16(
                false, a1, false, bf[j], (short)0, acc[1][j], false, false);
        }
    }

    // C/D layout: lane holds column n = (lane&15); VGPR r holds row m = r + 8*(lane>>4)
    const int col   = lane & 15;
    const int rowHi = (lane >> 4) * 8;
#pragma unroll
    for (int mi = 0; mi < 2; ++mi) {
#pragma unroll
        for (int j = 0; j < 4; ++j) {
#pragma unroll
            for (int r = 0; r < 8; ++r) {
                float v = acc[mi][j][r];
                if (TANH) v = fast_tanh(v);
                const int m = mTile + 16 * mi + rowHi + r;
                const int n = nTile + 16 * j + col;
                if (OUTF32) Df32[(size_t)b * sD32 + (size_t)m * ldd32 + n] = v;
                if (OUTBF)  Dbf [(size_t)b * sDbf + (size_t)m * lddbf + n] = (__bf16)v;
            }
        }
    }
}

// ---------------------------------------------------------------------------
// Masked softmax over SRC. One 256-thread block (8 wave32s) per (b,t) row.
// In-place f32 normalize in d_out's align region + bf16 copy of P for GEMM3.
// ---------------------------------------------------------------------------
__global__ __launch_bounds__(256)
void softmax_mask_kernel(float* __restrict__ align, __bf16* __restrict__ P,
                         const int* __restrict__ lens) {
    const int row = blockIdx.x;          // b*TGT + t
    const int b   = row / TGT;
    const int len = lens[b];
    float*  a = align + (size_t)row * SRC;
    __bf16* p = P     + (size_t)row * SRC;
    const int tid = threadIdx.x;

    __shared__ float red[8];

    float vals[4];
    float mx = -__builtin_inff();
#pragma unroll
    for (int j = 0; j < 4; ++j) {
        int s   = tid + 256 * j;
        float v = a[s];
        v = (s < len) ? v : -__builtin_inff();
        vals[j] = v;
        mx = fmaxf(mx, v);
    }
#pragma unroll
    for (int off = 16; off > 0; off >>= 1) mx = fmaxf(mx, __shfl_xor(mx, off, 32));
    if ((tid & 31) == 0) red[tid >> 5] = mx;
    __syncthreads();
#pragma unroll
    for (int w = 0; w < 8; ++w) mx = fmaxf(mx, red[w]);
    __syncthreads();

    float sum = 0.0f;
#pragma unroll
    for (int j = 0; j < 4; ++j) {
        float e = (vals[j] == -__builtin_inff()) ? 0.0f : __expf(vals[j] - mx);
        vals[j] = e;
        sum += e;
    }
#pragma unroll
    for (int off = 16; off > 0; off >>= 1) sum += __shfl_xor(sum, off, 32);
    if ((tid & 31) == 0) red[tid >> 5] = sum;
    __syncthreads();
    sum = 0.0f;
#pragma unroll
    for (int w = 0; w < 8; ++w) sum += red[w];

    const float inv = 1.0f / sum;
#pragma unroll
    for (int j = 0; j < 4; ++j) {
        int s   = tid + 256 * j;
        float r = vals[j] * inv;
        a[s] = r;
        p[s] = (__bf16)r;
    }
}

// ---------------------------------------------------------------------------
// Host launcher
// ---------------------------------------------------------------------------
extern "C" void kernel_launch(void* const* d_in, const int* in_sizes, int n_in,
                              void* d_out, int out_size, void* d_ws, size_t ws_size,
                              hipStream_t stream) {
    (void)in_sizes; (void)n_in; (void)out_size; (void)ws_size;

    const float* input  = (const float*)d_in[0];   // [B,TGT,DIM]
    const float* ctx    = (const float*)d_in[1];   // [B,SRC,DIM]
    const int*   lens   = (const int*)d_in[2];     // [B]
    const float* W_in   = (const float*)d_in[3];   // [DIM,DIM]
    const float* W_out  = (const float*)d_in[4];   // [DIM,2*DIM]

    float* attn_h = (float*)d_out;                                   // [B,TGT,DIM]
    float* align  = (float*)d_out + (size_t)BATCH * TGT * DIM;       // [B,TGT,SRC]

    // ---- workspace layout (bf16) ----
    char* ws = (char*)d_ws;
    const size_t nCtx  = (size_t)BATCH * SRC * DIM;      // 8M
    const size_t nWin  = (size_t)DIM * DIM;              // 1M
    const size_t nWout = (size_t)DIM * 2 * DIM;          // 2M
    const size_t nHt   = (size_t)BATCH * TGT * DIM;      // 8M
    const size_t nP    = (size_t)BATCH * TGT * SRC;      // 8M
    const size_t nCat  = (size_t)BATCH * TGT * 2 * DIM;  // 16M

    __bf16* ctx_bf  = (__bf16*)ws;                ws += nCtx  * 2;
    __bf16* Win_bf  = (__bf16*)ws;                ws += nWin  * 2;
    __bf16* Wout_bf = (__bf16*)ws;                ws += nWout * 2;
    __bf16* ht_bf   = (__bf16*)ws;                ws += nHt   * 2;
    __bf16* P_bf    = (__bf16*)ws;                ws += nP    * 2;
    __bf16* cat_bf  = (__bf16*)ws;                ws += nCat  * 2;   // [c | input] per row

    // ---- 1) precision conversion (one row per block, no int division) ----
    cvt_f32_bf16_rows<<<BATCH * SRC, 256, 0, stream>>>(ctx,   ctx_bf,  DIM,     DIM,     0);
    cvt_f32_bf16_rows<<<DIM,         256, 0, stream>>>(W_in,  Win_bf,  DIM,     DIM,     0);
    cvt_f32_bf16_rows<<<DIM,         256, 0, stream>>>(W_out, Wout_bf, 2 * DIM, 2 * DIM, 0);
    // input -> right half of the concat buffer (stride 2*DIM, offset DIM)
    cvt_f32_bf16_rows<<<BATCH * TGT, 256, 0, stream>>>(input, cat_bf,  DIM,     2 * DIM, DIM);

    const dim3 gGemm(DIM / 64, TGT / 32, BATCH);   // 16 x 32 x 8 waves
    const dim3 bWave(32, 1, 1);

    // ---- 2) h_t = input @ W_in^T   (A = concat right half, lda=2*DIM) ----
    wmma_gemm_bf16<true, false, false, true><<<gGemm, bWave, 0, stream>>>(
        cat_bf + DIM, (long long)TGT * 2 * DIM, 2 * DIM,
        Win_bf, 0LL, DIM,
        nullptr, 0LL, 0,
        ht_bf, (long long)TGT * DIM, DIM,
        /*K=*/DIM);

    // ---- 3) align = h_t @ context^T  -> f32 straight into d_out ----
    wmma_gemm_bf16<true, false, true, false><<<dim3(SRC / 64, TGT / 32, BATCH), bWave, 0, stream>>>(
        ht_bf, (long long)TGT * DIM, DIM,
        ctx_bf, (long long)SRC * DIM, DIM,
        align, (long long)TGT * SRC, SRC,
        nullptr, 0LL, 0,
        /*K=*/DIM);

    // ---- 4) masked softmax over SRC (in-place f32 + bf16 P) ----
    softmax_mask_kernel<<<BATCH * TGT, 256, 0, stream>>>(align, P_bf, lens);

    // ---- 5) c = P @ context  (B is N-contiguous) -> left half of concat ----
    wmma_gemm_bf16<false, false, false, true><<<gGemm, bWave, 0, stream>>>(
        P_bf, (long long)TGT * SRC, SRC,
        ctx_bf, (long long)SRC * DIM, DIM,
        nullptr, 0LL, 0,
        cat_bf, (long long)TGT * 2 * DIM, 2 * DIM,
        /*K=*/SRC);

    // ---- 6) attn_h = tanh(concat @ W_out^T), K = 2*DIM ----
    wmma_gemm_bf16<true, true, true, false><<<gGemm, bWave, 0, stream>>>(
        cat_bf, (long long)TGT * 2 * DIM, 2 * DIM,
        Wout_bf, 0LL, 2 * DIM,
        attn_h, (long long)TGT * DIM, DIM,
        nullptr, 0LL, 0,
        /*K=*/2 * DIM);
}
